// SimBlock_10471130268245
// MI455X (gfx1250) — matrix operand order repaired and checked by reading, not compiled, not run on hardware
//
#include <hip/hip_runtime.h>
#include <stdint.h>

// ---------------------------------------------------------------------------
// SimBlock (temporal similarity transform) for MI455X / gfx1250.
// All conv stages run as implicit GEMMs on v_wmma_f32_16x16x32_f16 (wave32),
// f16 operands + fp32 accumulate, BN+ReLU folded into epilogues.
// A tiles: [M][K] ld=40 halves (16B-aligned 8-half groups) -> 2x ds b128/frag.
// B tiles: [N][K] transposed ld=40 -> 2x ds b128/frag, conflict-free.
// u2/u3 weights packed [N][K] in global -> 2x 16B global loads per fragment.
// UV2C (u0..u3) fully fused per-sample-group in LDS (no global z0/z1/z2).
// Workspace requirement: ~80 MB.
// ---------------------------------------------------------------------------

typedef __attribute__((ext_vector_type(16))) _Float16 v16h;
typedef __attribute__((ext_vector_type(8)))  _Float16 v8h;
typedef __attribute__((ext_vector_type(2)))  _Float16 v2h;
typedef __attribute__((ext_vector_type(8)))  float    v8f;

#define DEV __device__ __forceinline__

DEV int lane_id() { return (int)(threadIdx.x & 31u); }

DEV v8f wmma16(v16h a, v16h b, v8f c) {
  // D = A(16x32 f16) * B(32x16 f16) + C(16x16 f32)
  return __builtin_amdgcn_wmma_f32_16x16x32_f16(false, a, false, b, (short)0, c,
                                                false, false);
}

// A fragment: 16x32 f16 tile, row-major, lda=40 halves (so every 8-half group
// is 16B aligned). lanes 0-15: m=0..15, K 0..7 / 16..23; lanes 16-31: K 8..15
// and 24..31.
DEV v16h load_a16(const _Float16* a) {
  int l = lane_id();
  const _Float16* p = a + (l & 15) * 40 + ((l >> 4) << 3);
  v8h lo = *(const v8h*)p;
  v8h hi = *(const v8h*)(p + 16);
  v16h r;
#pragma unroll
  for (int j = 0; j < 8; ++j) { r[j] = lo[j]; r[8 + j] = hi[j]; }
  return r;
}

// B fragment from transposed tile bT[N][K], ldbT=40 halves. Element j is
// B[k=(l>>4)*8+j (j<8) / 16+(l>>4)*8+(j-8) (j>=8)][n=l&15] (2x b128).
DEV v16h load_bT16(const _Float16* bT) {
  int l = lane_id();
  const _Float16* p = bT + (l & 15) * 40 + ((l >> 4) << 3);
  v8h lo = *(const v8h*)p;
  v8h hi = *(const v8h*)(p + 16);
  v16h r;
#pragma unroll
  for (int j = 0; j < 8; ++j) { r[j] = lo[j]; r[8 + j] = hi[j]; }
  return r;
}

// B fragment straight from a global [N][Krows] transposed weight matrix.
DEV v16h load_bT_g(const _Float16* bT, int nrow, int krows, int kcb) {
  int l = lane_id();
  const _Float16* p = bT + (nrow + (l & 15)) * krows + kcb + ((l >> 4) << 3);
  v8h lo = *(const v8h*)p;
  v8h hi = *(const v8h*)(p + 16);
  v16h r;
#pragma unroll
  for (int j = 0; j < 8; ++j) { r[j] = lo[j]; r[8 + j] = hi[j]; }
  return r;
}

// ---------------------------------------------------------------------------
// Small helper kernels
// ---------------------------------------------------------------------------

// BN -> affine: scale = g*rsqrt(v+eps); shift = b - m*scale.
__global__ void k_bn(const float* g, const float* b, const float* m,
                     const float* v, int C, float* aff) {
  int c = threadIdx.x;
  if (c < C) {
    float s = g[c] * rsqrtf(v[c] + 1e-5f);
    aff[c] = s;
    aff[64 + c] = b[c] - m[c] * s;
  }
}

__global__ void k_cvt(const float* __restrict__ in, _Float16* __restrict__ out,
                      int n) {
  int i = blockIdx.x * 256 + threadIdx.x;
  if (i < n) out[i] = (_Float16)in[i];
}

// Pack OIHW (O,I,3,3) fp32 weights into transposed GEMM-B layout [O][Krows]
// f16, k = i*9 + ky*3 + kx ; columns k >= I*9 are zero-padding.
__global__ void k_packT(const float* __restrict__ w, int O, int I, int Krows,
                        _Float16* __restrict__ BT) {
  int idx = blockIdx.x * 256 + threadIdx.x;
  if (idx >= O * Krows) return;
  int o = idx / Krows, k = idx % Krows;
  float v = 0.f;
  if (k < I * 9) v = w[(o * I + k / 9) * 9 + k % 9];
  BT[idx] = (_Float16)v;
}

// up_w (64,256) [c][d] -> BupT [d][c] f16.
__global__ void k_pack_upT(const float* __restrict__ w,
                           _Float16* __restrict__ BT) {
  int idx = blockIdx.x * 256 + threadIdx.x;
  if (idx >= 256 * 64) return;
  int d = idx >> 6, c = idx & 63;
  BT[idx] = (_Float16)w[c * 256 + d];
}

// ---------------------------------------------------------------------------
// Stage 1: downsample conv 256->64, 3x3 stride 2, BN+ReLU. Implicit GEMM:
// M=12544 (bt,oh,ow), K=2304 (ic,ky,kx), N=64. Writes y fp32 (16,64,28,28).
// ---------------------------------------------------------------------------
__global__ __launch_bounds__(128) void k_gemm_ds(
    const _Float16* __restrict__ xh, const _Float16* __restrict__ BdsT,
    const float* __restrict__ bnaff, float* __restrict__ y) {
  __shared__ __align__(16) _Float16 As[64 * 40];
  __shared__ __align__(16) _Float16 BsT[64 * 40];
  const int tid = threadIdx.x;
  const int wv = tid >> 5;
  const int mbase = blockIdx.x * 64;
  const int kp = (tid & 15) << 1;  // this thread stages k-pair kp,kp+1
  const int rb = tid >> 4;         // row group 0..7
  int bt_[8], ih0_[8], iw0_[8];
#pragma unroll
  for (int idx = 0; idx < 8; ++idx) {
    int m = mbase + rb + idx * 8;
    int bt = m / 784, pos = m % 784;
    bt_[idx] = bt * 256 * 3136;
    ih0_[idx] = (pos / 28) * 2 - 1;
    iw0_[idx] = (pos % 28) * 2 - 1;
  }
  v8f acc0 = {}, acc1 = {}, acc2 = {}, acc3 = {};
  for (int kc = 0; kc < 2304; kc += 32) {
    int k0 = kc + kp, k1 = k0 + 1;
    int ic0 = k0 / 9, r0 = k0 % 9, ky0 = r0 / 3, kx0 = r0 % 3;
    int ic1 = k1 / 9, r1 = k1 % 9, ky1 = r1 / 3, kx1 = r1 % 3;
#pragma unroll
    for (int idx = 0; idx < 8; ++idx) {
      int ihA = ih0_[idx] + ky0, iwA = iw0_[idx] + kx0;
      int ihB = ih0_[idx] + ky1, iwB = iw0_[idx] + kx1;
      _Float16 h0 = (_Float16)0.f, h1 = (_Float16)0.f;
      if ((unsigned)ihA < 56u && (unsigned)iwA < 56u)
        h0 = xh[bt_[idx] + ic0 * 3136 + ihA * 56 + iwA];
      if ((unsigned)ihB < 56u && (unsigned)iwB < 56u)
        h1 = xh[bt_[idx] + ic1 * 3136 + ihB * 56 + iwB];
      v2h pk; pk[0] = h0; pk[1] = h1;
      *(v2h*)&As[(rb + idx * 8) * 40 + kp] = pk;
    }
#pragma unroll
    for (int c2 = 0; c2 < 2; ++c2) {
      int cc = tid + c2 * 128;               // 0..255 -> 64 n x 4 groups
      int n = cc >> 2, ko = (cc & 3) << 3;
      *(v8h*)&BsT[n * 40 + ko] = *(const v8h*)&BdsT[n * 2304 + kc + ko];
      if (kc + 32 < 2304)
        __builtin_prefetch(&BdsT[n * 2304 + kc + 32 + ko], 0, 3);
    }
    __syncthreads();
    v16h af = load_a16(&As[(wv * 16) * 40]);
    acc0 = wmma16(af, load_bT16(&BsT[0 * 40]), acc0);
    acc1 = wmma16(af, load_bT16(&BsT[16 * 40]), acc1);
    acc2 = wmma16(af, load_bT16(&BsT[32 * 40]), acc2);
    acc3 = wmma16(af, load_bT16(&BsT[48 * 40]), acc3);
    __syncthreads();
  }
  const float* sc = bnaff;
  const float* sh = bnaff + 64;
  int l = lane_id();
  int nl = l & 15, mo = (l >> 4) << 3;
#pragma unroll
  for (int nt = 0; nt < 4; ++nt) {
    v8f acc = (nt == 0) ? acc0 : (nt == 1) ? acc1 : (nt == 2) ? acc2 : acc3;
#pragma unroll
    for (int r = 0; r < 8; ++r) {
      int m = mbase + wv * 16 + mo + r;
      int n = nt * 16 + nl;
      int bt = m / 784, pos = m % 784;
      float v = fmaxf(acc[r] * sc[n] + sh[n], 0.f);
      y[(bt * 64 + n) * 784 + pos] = v;
    }
  }
}

// ---------------------------------------------------------------------------
// Stage 2a: channel l2norm per spatial position (x / sqrt(sum_c x^2 + 1e-6)).
// ---------------------------------------------------------------------------
__global__ void k_l2norm(const float* __restrict__ y, float* __restrict__ yn) {
  int p = blockIdx.x * 256 + threadIdx.x;
  if (p >= 12544) return;
  int bt = p / 784, pos = p % 784;
  const float* src = y + bt * 64 * 784 + pos;
  float s = 1e-6f;
#pragma unroll 8
  for (int c = 0; c < 64; ++c) { float v = src[c * 784]; s += v * v; }
  float inv = rsqrtf(s);
  float* dst = yn + bt * 64 * 784 + pos;
#pragma unroll 8
  for (int c = 0; c < 64; ++c) dst[c * 784] = src[c * 784] * inv;
}

// ---------------------------------------------------------------------------
// Stage 2b: windowed temporal-spatial correlation directly into the
// (b,t,h,w, l,u,v) layout consumed by UV2C. sim[(s)*405 + l*81+u*9+v].
// ---------------------------------------------------------------------------
__global__ void k_corr(const float* __restrict__ yn, float* __restrict__ sim) {
  int idx = blockIdx.x * 256 + threadIdx.x;
  if (idx >= 12544 * 405) return;
  int s = idx / 405, r = idx % 405;
  int l = r / 81, uv = r % 81, u = uv / 9, v = uv % 9;
  int b = s / 6272, rem = s % 6272;
  int t = rem / 784, pos = rem % 784;
  int i = pos / 28, j = pos % 28;
  float acc = 0.f;
  int f = t + l - 2;
  int ii = i + u - 4, jj = j + v - 4;
  if (f >= 0 && f < 8 && (unsigned)ii < 28u && (unsigned)jj < 28u) {
    const float* f1 = yn + ((b * 8 + t) * 64) * 784 + pos;
    const float* f2 = yn + ((b * 8 + f) * 64) * 784 + ii * 28 + jj;
#pragma unroll 8
    for (int c = 0; c < 64; ++c) acc += f1[c * 784] * f2[c * 784];
  }
  sim[idx] = acc;
}

// ---------------------------------------------------------------------------
// Stage 3: fused UV2C (u0 -> u1 -> u2 -> u3), 4 samples per 256-thread WG.
// LDS arena (aliased regions, peak 41.3 KB):
//   z0S  f32 [4][4][405]   @ 0      (u0 out / u1 in)
//   simS f32 [4][405]      @ 25920  (u0 in)
//   z1h  f16 [4][16][125]  @ 25920  (u1 out / u2 in)   (simS dead)
//   z2h  f16 [4][64][45]   @ 0      (u2 out / u3 in)   (z0S dead)
// u2: WMMA GEMM M=180(pad 192) K=144(pad 160) N=64;
// u3: WMMA GEMM M=20(pad 32)  K=576          N=64.
// Output zt f16 (b*5, 64, 8, 28, 28).
// ---------------------------------------------------------------------------
DEV v16h u2_a_frag(const _Float16* z1h, int mt, int kcb) {
  int l = lane_id();
  int row = mt * 16 + (l & 15);
  bool rv = row < 180;
  int samp = row / 45, p = row % 45;
  int ll = p / 9, oy = (p % 9) / 3, ox = p % 3;
  int kb = (l >> 4) << 3;
  v16h a;
#pragma unroll
  for (int j = 0; j < 16; ++j) {
    int k = kcb + ((j < 8) ? (kb + j) : (16 + kb + (j - 8)));
    _Float16 v = (_Float16)0.f;
    if (rv && k < 144) {
      int ic = k / 9, r9 = k % 9;
      int yy = oy * 2 - 1 + r9 / 3, xx = ox * 2 - 1 + r9 % 3;
      if ((unsigned)yy < 5u && (unsigned)xx < 5u)
        v = z1h[samp * 2000 + ic * 125 + ll * 25 + yy * 5 + xx];
    }
    a[j] = v;
  }
  return a;
}

__global__ __launch_bounds__(256) void k_uv2c(
    const float* __restrict__ sim, const float* __restrict__ w0,
    const float* __restrict__ w1, const _Float16* __restrict__ Bu2T,
    const _Float16* __restrict__ Bu3T, const float* __restrict__ bnaff,
    _Float16* __restrict__ zt) {
  __shared__ __align__(16) char ldsbuf[42304];
  float* z0S = (float*)ldsbuf;
  float* simS = (float*)(ldsbuf + 25920);
  _Float16* z1h = (_Float16*)(ldsbuf + 25920);
  _Float16* z2h = (_Float16*)ldsbuf;
  const int tid = threadIdx.x;
  const int wave = tid >> 5;
  const int wg = blockIdx.x;

  // P0: stage sim for 4 samples
  for (int e = tid; e < 4 * 405; e += 256) simS[e] = sim[wg * 4 * 405 + e];
  __syncthreads();

  // P1: u0 (1->4 ch, 3x3 over 9x9, pad 1) + BN + ReLU  (VALU)
  {
    const float* sc = bnaff + 1 * 128;
    const float* sh = sc + 64;
    for (int e = tid; e < 4 * 4 * 405; e += 256) {
      int samp = e / 1620, rem = e % 1620;
      int oc = rem / 405, p = rem % 405;
      int l = p / 81, uv = p % 81, u = uv / 9, v = uv % 9;
      float acc = 0.f;
#pragma unroll
      for (int ky = 0; ky < 3; ++ky) {
        int uu = u + ky - 1;
        if ((unsigned)uu >= 9u) continue;
#pragma unroll
        for (int kx = 0; kx < 3; ++kx) {
          int vv = v + kx - 1;
          if ((unsigned)vv >= 9u) continue;
          acc += w0[oc * 9 + ky * 3 + kx] * simS[samp * 405 + l * 81 + uu * 9 + vv];
        }
      }
      z0S[samp * 1620 + oc * 405 + p] = fmaxf(acc * sc[oc] + sh[oc], 0.f);
    }
  }
  __syncthreads();

  // P2: u1 (4->16 ch, 3x3 stride 2, pad 1: 9x9 -> 5x5) + BN + ReLU (VALU)
  {
    const float* sc = bnaff + 2 * 128;
    const float* sh = sc + 64;
    for (int e = tid; e < 4 * 16 * 125; e += 256) {
      int samp = e / 2000, rem = e % 2000;
      int oc = rem / 125, p = rem % 125;
      int l = p / 25, yy = (p % 25) / 5, xx = p % 5;
      float acc = 0.f;
      for (int ic = 0; ic < 4; ++ic) {
#pragma unroll
        for (int ky = 0; ky < 3; ++ky) {
          int iy = yy * 2 - 1 + ky;
          if ((unsigned)iy >= 9u) continue;
#pragma unroll
          for (int kx = 0; kx < 3; ++kx) {
            int ix = xx * 2 - 1 + kx;
            if ((unsigned)ix >= 9u) continue;
            acc += w1[(oc * 4 + ic) * 9 + ky * 3 + kx] *
                   z0S[samp * 1620 + ic * 405 + l * 81 + iy * 9 + ix];
          }
        }
      }
      z1h[samp * 2000 + oc * 125 + p] = (_Float16)fmaxf(acc * sc[oc] + sh[oc], 0.f);
    }
  }
  __syncthreads();

  // P3: u2 (16->64 ch, 3x3 stride 2: 5x5 -> 3x3) via WMMA. 12 Mtiles x 4 Ntiles.
  {
    const float* sc = bnaff + 3 * 128;
    const float* sh = sc + 64;
    for (int task = wave; task < 48; task += 8) {
      int mt = task >> 2, nt = task & 3;
      v8f acc = {};
#pragma unroll
      for (int kc = 0; kc < 5; ++kc) {
        v16h a = u2_a_frag(z1h, mt, kc * 32);
        v16h bf = load_bT_g(Bu2T, nt * 16, 160, kc * 32);
        acc = wmma16(a, bf, acc);
      }
      int l = lane_id();
      int n = nt * 16 + (l & 15);
      int mo = (l >> 4) << 3;
#pragma unroll
      for (int r = 0; r < 8; ++r) {
        int row = mt * 16 + mo + r;
        if (row < 180) {
          int samp = row / 45, p = row % 45;
          float v = fmaxf(acc[r] * sc[n] + sh[n], 0.f);
          z2h[samp * 2880 + n * 45 + p] = (_Float16)v;
        }
      }
    }
  }
  __syncthreads();

  // P4: u3 (64->64 ch, 3x3 valid: 3x3 -> 1x1) via WMMA. 2 Mtiles x 4 Ntiles.
  {
    const float* sc = bnaff + 4 * 128;
    const float* sh = sc + 64;
    int mt = wave >> 2, nt = wave & 3;
    v8f acc = {};
    for (int kc = 0; kc < 18; ++kc) {
      int kcb = kc * 32;
      int l = lane_id();
      v16h a;
      int row = mt * 16 + (l & 15);
      bool rv = row < 20;
      int samp = rv ? row / 5 : 0, ll = rv ? row % 5 : 0;
      int kb = (l >> 4) << 3;
#pragma unroll
      for (int j = 0; j < 16; ++j) {
        int k = kcb + ((j < 8) ? (kb + j) : (16 + kb + (j - 8)));
        _Float16 v = (_Float16)0.f;
        if (rv) {
          int ic = k / 9, r9 = k % 9;
          v = z2h[samp * 2880 + ic * 45 + ll * 9 + r9];
        }
        a[j] = v;
      }
      v16h bf = load_bT_g(Bu3T, nt * 16, 576, kcb);
      acc = wmma16(a, bf, acc);
    }
    int l = lane_id();
    int n = nt * 16 + (l & 15);
    int mo = (l >> 4) << 3;
#pragma unroll
    for (int r = 0; r < 8; ++r) {
      int row = mt * 16 + mo + r;
      if (row < 20) {
        int samp = row / 5, ll = row % 5;
        int gs = wg * 4 + samp;                       // global sample (b,t,i,j)
        int b = gs / 6272, rem = gs % 6272;
        int t = rem / 784, pos = rem % 784;
        float v = fmaxf(acc[r] * sc[n] + sh[n], 0.f);
        zt[((b * 5 + ll) * 64 + n) * 6272 + t * 784 + pos] = (_Float16)v;
      }
    }
  }
}

// ---------------------------------------------------------------------------
// Stage 4: spatial 3x3 conv (1,3,3) pad (0,1,1) GEMM kernel (s1, s2, s3).
// in: (B, Cin, 8, 28, 28) f16 ; out: (B, 64, 8, 28, 28) f16 ; M = B*8*784.
// ---------------------------------------------------------------------------
__global__ __launch_bounds__(128) void k_conv_s(
    const _Float16* __restrict__ in, int Cin, int kTotal,
    const _Float16* __restrict__ BwT, const float* __restrict__ bnaff,
    _Float16* __restrict__ out) {
  __shared__ __align__(16) _Float16 As[64 * 40];
  __shared__ __align__(16) _Float16 BsT[64 * 40];
  const int tid = threadIdx.x;
  const int wv = tid >> 5;
  const int mbase = blockIdx.x * 64;
  const int kp = (tid & 15) << 1;
  const int rb = tid >> 4;
  int ibase_[8], i0_[8], j0_[8];
#pragma unroll
  for (int idx = 0; idx < 8; ++idx) {
    int m = mbase + rb + idx * 8;
    int bb = m / 6272, rem = m % 6272;
    int t = rem / 784, pos = rem % 784;
    ibase_[idx] = (bb * Cin * 8 + t) * 784;
    i0_[idx] = pos / 28 - 1;
    j0_[idx] = pos % 28 - 1;
  }
  v8f acc0 = {}, acc1 = {}, acc2 = {}, acc3 = {};
  for (int kc = 0; kc < kTotal; kc += 32) {
    int k0 = kc + kp, k1 = k0 + 1;
    int ic0 = k0 / 9, r0 = k0 % 9, ky0 = r0 / 3, kx0 = r0 % 3;
    int ic1 = k1 / 9, r1 = k1 % 9, ky1 = r1 / 3, kx1 = r1 % 3;
#pragma unroll
    for (int idx = 0; idx < 8; ++idx) {
      int iA = i0_[idx] + ky0, jA = j0_[idx] + kx0;
      int iB = i0_[idx] + ky1, jB = j0_[idx] + kx1;
      _Float16 h0 = (_Float16)0.f, h1 = (_Float16)0.f;
      if ((unsigned)iA < 28u && (unsigned)jA < 28u)
        h0 = in[ibase_[idx] + ic0 * 6272 + iA * 28 + jA];
      if ((unsigned)iB < 28u && (unsigned)jB < 28u)
        h1 = in[ibase_[idx] + ic1 * 6272 + iB * 28 + jB];
      v2h pk; pk[0] = h0; pk[1] = h1;
      *(v2h*)&As[(rb + idx * 8) * 40 + kp] = pk;
    }
#pragma unroll
    for (int c2 = 0; c2 < 2; ++c2) {
      int cc = tid + c2 * 128;
      int n = cc >> 2, ko = (cc & 3) << 3;
      *(v8h*)&BsT[n * 40 + ko] = *(const v8h*)&BwT[n * kTotal + kc + ko];
      if (kc + 32 < kTotal)
        __builtin_prefetch(&BwT[n * kTotal + kc + 32 + ko], 0, 3);
    }
    __syncthreads();
    v16h af = load_a16(&As[(wv * 16) * 40]);
    acc0 = wmma16(af, load_bT16(&BsT[0 * 40]), acc0);
    acc1 = wmma16(af, load_bT16(&BsT[16 * 40]), acc1);
    acc2 = wmma16(af, load_bT16(&BsT[32 * 40]), acc2);
    acc3 = wmma16(af, load_bT16(&BsT[48 * 40]), acc3);
    __syncthreads();
  }
  const float* sc = bnaff;
  const float* sh = bnaff + 64;
  int l = lane_id();
  int nl = l & 15, mo = (l >> 4) << 3;
#pragma unroll
  for (int nt = 0; nt < 4; ++nt) {
    v8f acc = (nt == 0) ? acc0 : (nt == 1) ? acc1 : (nt == 2) ? acc2 : acc3;
#pragma unroll
    for (int r = 0; r < 8; ++r) {
      int m = mbase + wv * 16 + mo + r;
      int n = nt * 16 + nl;
      int bb = m / 6272, rem = m % 6272;
      int t = rem / 784, pos = rem % 784;
      float v = fmaxf(acc[r] * sc[n] + sh[n], 0.f);
      out[((bb * 64 + n) * 8 + t) * 784 + pos] = (_Float16)v;
    }
  }
}

// ---------------------------------------------------------------------------
// Stage 5a: out = relu(x) everywhere (odd positions of the stride-2 scatter).
// ---------------------------------------------------------------------------
__global__ void k_fill(const float* __restrict__ x, float* __restrict__ out,
                       int n) {
  int i = blockIdx.x * 256 + threadIdx.x;
  if (i < n) out[i] = fmaxf(x[i], 0.f);
}

// ---------------------------------------------------------------------------
// Stage 5b: up-projection GEMM M=12544 K=64 N=256, scattered to even/even
// positions with residual + ReLU. grid (196, 4).
// ---------------------------------------------------------------------------
__global__ __launch_bounds__(128) void k_up(const _Float16* __restrict__ s3o,
                                            const _Float16* __restrict__ BupT,
                                            const float* __restrict__ x,
                                            float* __restrict__ out) {
  __shared__ __align__(16) _Float16 As[64 * 40];
  __shared__ __align__(16) _Float16 BsT[64 * 40];
  const int tid = threadIdx.x;
  const int wv = tid >> 5;
  const int mbase = blockIdx.x * 64;
  const int nbase = blockIdx.y * 64;
  const int kp = (tid & 15) << 1;
  const int rb = tid >> 4;
  int abase_[8];
#pragma unroll
  for (int idx = 0; idx < 8; ++idx) {
    int m = mbase + rb + idx * 8;
    int b = m / 6272, rem = m % 6272;
    int t = rem / 784, pos = rem % 784;
    abase_[idx] = (b * 512 + t) * 784 + pos;  // + c*6272
  }
  v8f acc0 = {}, acc1 = {}, acc2 = {}, acc3 = {};
  for (int kc = 0; kc < 64; kc += 32) {
    int c0 = kc + kp, c1 = c0 + 1;
#pragma unroll
    for (int idx = 0; idx < 8; ++idx) {
      v2h pk;
      pk[0] = s3o[abase_[idx] + c0 * 6272];
      pk[1] = s3o[abase_[idx] + c1 * 6272];
      *(v2h*)&As[(rb + idx * 8) * 40 + kp] = pk;
    }
#pragma unroll
    for (int c2 = 0; c2 < 2; ++c2) {
      int cc = tid + c2 * 128;
      int n = cc >> 2, ko = (cc & 3) << 3;
      *(v8h*)&BsT[n * 40 + ko] = *(const v8h*)&BupT[(nbase + n) * 64 + kc + ko];
    }
    __syncthreads();
    v16h af = load_a16(&As[(wv * 16) * 40]);
    acc0 = wmma16(af, load_bT16(&BsT[0 * 40]), acc0);
    acc1 = wmma16(af, load_bT16(&BsT[16 * 40]), acc1);
    acc2 = wmma16(af, load_bT16(&BsT[32 * 40]), acc2);
    acc3 = wmma16(af, load_bT16(&BsT[48 * 40]), acc3);
    __syncthreads();
  }
  int l = lane_id();
  int nl = l & 15, mo = (l >> 4) << 3;
#pragma unroll
  for (int nt = 0; nt < 4; ++nt) {
    v8f acc = (nt == 0) ? acc0 : (nt == 1) ? acc1 : (nt == 2) ? acc2 : acc3;
#pragma unroll
    for (int r = 0; r < 8; ++r) {
      int m = mbase + wv * 16 + mo + r;
      int d = nbase + nt * 16 + nl;
      int b = m / 6272, rem = m % 6272;
      int t = rem / 784, pos = rem % 784;
      int i = pos / 28, j = pos % 28;
      int oidx = ((b * 8 + t) * 256 + d) * 3136 + (i * 2) * 56 + j * 2;
      out[oidx] = fmaxf(x[oidx] + acc[r], 0.f);
    }
  }
}

// ---------------------------------------------------------------------------
// Host launcher
// ---------------------------------------------------------------------------
extern "C" void kernel_launch(void* const* d_in, const int* in_sizes, int n_in,
                              void* d_out, int out_size, void* d_ws,
                              size_t ws_size, hipStream_t stream) {
  (void)n_in; (void)out_size; (void)ws_size;
  const float *x = nullptr, *ds_w = nullptr, *up_w = nullptr;
  const float *u0_w = nullptr, *u1_w = nullptr, *u2_w = nullptr, *u3_w = nullptr;
  const float *s1_w = nullptr, *s2_w = nullptr, *s3_w = nullptr;
  // bn stage order: 0=ds 1=u0 2=u1 3=u2 4=u3 5=s1 6=s2 7=s3
  const float *bg[8], *bb[8], *bm[8], *bv[8];
  const int NX = 16 * 256 * 56 * 56;  // 12,845,056

  bool ins = (in_sizes[0] == NX);  // insertion-order flatten (x first)
  auto F = [&](int i) { return (const float*)d_in[i]; };
  if (ins) {
    // x, ds_w, ds_bn{g,b,m,v}, u0_w, u0_bn, u1_w, u1_bn, u2_w, u2_bn,
    // u3_w, u3_bn, s1_w, s1_bn, s2_w, s2_bn, s3_w, s3_bn, up_w
    x = F(0); ds_w = F(1);
    auto BN = [&](int st, int base) { bg[st]=F(base); bb[st]=F(base+1); bm[st]=F(base+2); bv[st]=F(base+3); };
    BN(0, 2);
    u0_w = F(6);  BN(1, 7);
    u1_w = F(11); BN(2, 12);
    u2_w = F(16); BN(3, 17);
    u3_w = F(21); BN(4, 22);
    s1_w = F(26); BN(5, 27);
    s2_w = F(31); BN(6, 32);
    s3_w = F(36); BN(7, 37);
    up_w = F(41);
  } else {
    // jax pytree sorted: params{ds_bn{b,g,m,v},ds_w,s1_bn,s1_w,s2_bn,s2_w,
    // s3_bn,s3_w,u0_bn,u0_w,u1_bn,u1_w,u2_bn,u2_w,u3_bn,u3_w,up_w}, x
    auto BN = [&](int st, int base) { bb[st]=F(base); bg[st]=F(base+1); bm[st]=F(base+2); bv[st]=F(base+3); };
    BN(0, 0);  ds_w = F(4);
    BN(5, 5);  s1_w = F(9);
    BN(6, 10); s2_w = F(14);
    BN(7, 15); s3_w = F(19);
    BN(1, 20); u0_w = F(24);
    BN(2, 25); u1_w = F(29);
    BN(3, 30); u2_w = F(34);
    BN(4, 35); u3_w = F(39);
    up_w = F(40);
    x = F(41);
  }

  // ---- carve workspace (~80 MB) ----
  uintptr_t wp = (uintptr_t)d_ws;
  auto carve = [&](size_t bytes) -> void* {
    void* p = (void*)wp;
    wp += (bytes + 255) & ~(size_t)255;
    return p;
  };
  float* bnaff    = (float*)carve(8 * 128 * sizeof(float));
  _Float16* xh    = (_Float16*)carve((size_t)NX * 2);
  _Float16* BdsT  = (_Float16*)carve((size_t)64 * 2304 * 2);
  _Float16* Bu2T  = (_Float16*)carve((size_t)64 * 160 * 2);  // K padded ->160
  _Float16* Bu3T  = (_Float16*)carve((size_t)64 * 576 * 2);
  _Float16* Bs1T  = (_Float16*)carve((size_t)64 * 576 * 2);
  _Float16* Bs2T  = (_Float16*)carve((size_t)64 * 576 * 2);
  _Float16* Bs3T  = (_Float16*)carve((size_t)64 * 2880 * 2);
  _Float16* BupT  = (_Float16*)carve((size_t)256 * 64 * 2);
  float* y        = (float*)carve((size_t)802816 * 4);
  float* yn       = (float*)carve((size_t)802816 * 4);
  float* sim      = (float*)carve((size_t)5080320 * 4);
  _Float16* zt    = (_Float16*)carve((size_t)4014080 * 2);
  _Float16* s1o   = (_Float16*)carve((size_t)4014080 * 2);
  _Float16* s2o   = (_Float16*)carve((size_t)4014080 * 2);
  _Float16* s3o   = (_Float16*)carve((size_t)802816 * 2);

  // ---- preprocessing ----
  static const int Cst[8] = {64, 4, 16, 64, 64, 64, 64, 64};
  for (int st = 0; st < 8; ++st)
    k_bn<<<1, 64, 0, stream>>>(bg[st], bb[st], bm[st], bv[st], Cst[st],
                               bnaff + st * 128);
  k_cvt<<<(NX + 255) / 256, 256, 0, stream>>>(x, xh, NX);
  k_packT<<<(64 * 2304 + 255) / 256, 256, 0, stream>>>(ds_w, 64, 256, 2304, BdsT);
  k_packT<<<(64 * 160 + 255) / 256, 256, 0, stream>>>(u2_w, 64, 16, 160, Bu2T);
  k_packT<<<(64 * 576 + 255) / 256, 256, 0, stream>>>(u3_w, 64, 64, 576, Bu3T);
  k_packT<<<(64 * 576 + 255) / 256, 256, 0, stream>>>(s1_w, 64, 64, 576, Bs1T);
  k_packT<<<(64 * 576 + 255) / 256, 256, 0, stream>>>(s2_w, 64, 64, 576, Bs2T);
  k_packT<<<(64 * 2880 + 255) / 256, 256, 0, stream>>>(s3_w, 64, 320, 2880, Bs3T);
  k_pack_upT<<<(256 * 64 + 255) / 256, 256, 0, stream>>>(up_w, BupT);

  // ---- pipeline ----
  k_gemm_ds<<<196, 128, 0, stream>>>(xh, BdsT, bnaff + 0 * 128, y);
  k_l2norm<<<(12544 + 255) / 256, 256, 0, stream>>>(y, yn);
  k_corr<<<(12544 * 405 + 255) / 256, 256, 0, stream>>>(yn, sim);
  k_uv2c<<<3136, 256, 0, stream>>>(sim, u0_w, u1_w, Bu2T, Bu3T, bnaff, zt);
  k_conv_s<<<980, 128, 0, stream>>>(zt, 64, 576, Bs1T, bnaff + 5 * 128, s1o);
  k_conv_s<<<980, 128, 0, stream>>>(s1o, 64, 576, Bs2T, bnaff + 6 * 128, s2o);
  k_conv_s<<<196, 128, 0, stream>>>(s2o, 320, 2880, Bs3T, bnaff + 7 * 128, s3o);
  k_fill<<<(NX + 255) / 256, 256, 0, stream>>>(x, (float*)d_out, NX);
  k_up<<<dim3(196, 4), 128, 0, stream>>>(s3o, BupT, x, (float*)d_out);
}
